// TransNeXt_24799141167818
// MI455X (gfx1250) — matrix operand order repaired and checked by reading, not compiled, not run on hardware
//
#include <hip/hip_runtime.h>
#include <hip/hip_bf16.h>

typedef __attribute__((ext_vector_type(16))) __bf16 v16bf;
typedef __attribute__((ext_vector_type(8)))  __bf16 v8bf;
typedef __attribute__((ext_vector_type(8)))  float  v8f;
typedef __attribute__((ext_vector_type(2)))  float  v2f;

#define NTOK 4096      // N = 64*64
#define CDIM 128
#define NHEAD 4
#define HD 32
#define PL 64
#define LL 9
#define NEG_INF (-3.0e38f)

// ---------------- weight packing: Wq|Wkv|Wsr -> bf16 (512,128), bias (512), Wproj -> bf16 ----
__global__ __launch_bounds__(256) void pack_w(
    const float* __restrict__ Wq, const float* __restrict__ Wkv, const float* __restrict__ Wsr,
    const float* __restrict__ bq, const float* __restrict__ bkv, const float* __restrict__ bsr,
    const float* __restrict__ Wproj,
    __bf16* __restrict__ Wc, float* __restrict__ bc, __bf16* __restrict__ Wp)
{
    int i = blockIdx.x * 256 + threadIdx.x;      // 0..65535
    int o = i >> 7, k = i & 127;
    float wv;
    if (o < 128)      wv = Wq [o * 128 + k];
    else if (o < 384) wv = Wkv[(o - 128) * 128 + k];
    else              wv = Wsr[(o - 384) * 128 + k];
    Wc[i] = (__bf16)wv;
    if (i < 16384) Wp[i] = (__bf16)Wproj[i];
    if (i < 512) {
        float bv;
        if (i < 128)      bv = bq [i];
        else if (i < 384) bv = bkv[i - 128];
        else              bv = bsr[i - 384];
        bc[i] = bv;
    }
}

// ---------------- f32 -> bf16 elementwise --------------------------------------------------
__global__ __launch_bounds__(256) void cvt_bf16(const float* __restrict__ src,
                                                __bf16* __restrict__ dst, int n)
{
    for (int i = blockIdx.x * 256 + threadIdx.x; i < n; i += gridDim.x * 256)
        dst[i] = (__bf16)src[i];
}

// ---------------- CPB MLP bias table: t_tab[h*T + i] ---------------------------------------
__global__ __launch_bounds__(256) void cpb_table(
    const float* __restrict__ tab, const float* __restrict__ w1, const float* __restrict__ b1,
    const float* __restrict__ w2, const float* __restrict__ b2,
    float* __restrict__ ttab, int T)
{
    int i = blockIdx.x;
    int tid = threadIdx.x;
    float c0 = tab[i * 2 + 0], c1 = tab[i * 2 + 1];
    float part0 = 0.f, part1 = 0.f, part2 = 0.f, part3 = 0.f;
    for (int j = tid; j < 512; j += 256) {
        float hdn = fmaxf(c0 * w1[j * 2] + c1 * w1[j * 2 + 1] + b1[j], 0.f);
        part0 += hdn * w2[0 * 512 + j];
        part1 += hdn * w2[1 * 512 + j];
        part2 += hdn * w2[2 * 512 + j];
        part3 += hdn * w2[3 * 512 + j];
    }
    __shared__ float red[4][256];
    red[0][tid] = part0; red[1][tid] = part1; red[2][tid] = part2; red[3][tid] = part3;
    __syncthreads();
    for (int off = 128; off; off >>= 1) {
        if (tid < off)
            for (int h = 0; h < 4; ++h) red[h][tid] += red[h][tid + off];
        __syncthreads();
    }
    if (tid < 4) ttab[(size_t)tid * T + i] = red[tid][0] + b2[tid];
}

// ---------------- generic bf16 WMMA GEMM: Y = A(MxK) @ W(NcolsxK)^T + bias ------------------
// MODE 0: fused qkv+sr epilogue (o0=q_raw (B,H,N,32), o1=k_raw, o2=v, o3=xs (M,128) w/ GELU)
// MODE 1: pool kv epilogue     (o0=kp_raw (B,H,64,32), o1=vp)
// MODE 2: plain row-major      (o0=(M,Ncols))
template <int MODE>
__global__ __launch_bounds__(128) void gemm_bf16(
    const __bf16* __restrict__ A, const __bf16* __restrict__ W,
    const float* __restrict__ bias,
    float* __restrict__ o0, float* __restrict__ o1,
    float* __restrict__ o2, float* __restrict__ o3,
    int M, int Ncols, int K)
{
    int wid = (blockIdx.x * 128 + threadIdx.x) >> 5;
    int tiles_n = Ncols >> 4;
    int m0 = (wid / tiles_n) << 4;
    int n0 = (wid % tiles_n) << 4;
    int lane = threadIdx.x & 31;
    int l = lane & 15, hi = lane >> 4;

    // rows are 16-byte aligned (K multiple of 8, base 256B aligned) -> b128 loads
    const v8bf* A8 = (const v8bf*)(A + (size_t)(m0 + l) * K);
    const v8bf* W8 = (const v8bf*)(W + (size_t)(n0 + l) * K);

    v8f acc = {};
    for (int k0 = 0; k0 < K; k0 += 32) {
        int c = k0 >> 3;
        // A: documented 16-bit 16x32 layout (lane half hi: K in [8hi..8hi+7] U [16+8hi..23+8hi])
        v8bf a0 = A8[c + hi];
        v8bf a1 = A8[c + 2 + hi];
        // B: same layout as A (lane half hi: K in [8hi..8hi+7] U [16+8hi..23+8hi])
        v8bf b0 = W8[c + hi];
        v8bf b1 = W8[c + 2 + hi];
        v16bf av = __builtin_shufflevector(a0, a1, 0, 1, 2, 3, 4, 5, 6, 7,
                                                   8, 9, 10, 11, 12, 13, 14, 15);
        v16bf bv = __builtin_shufflevector(b0, b1, 0, 1, 2, 3, 4, 5, 6, 7,
                                                   8, 9, 10, 11, 12, 13, 14, 15);
        acc = __builtin_amdgcn_wmma_f32_16x16x32_bf16(false, av, false, bv,
                                                      (short)0, acc, false, false);
    }

    int col = n0 + l;
    float bcol = bias[col];
#pragma unroll
    for (int i = 0; i < 8; ++i) {
        int m = m0 + i + hi * 8;
        float yv = acc[i] + bcol;
        if (MODE == 0) {
            int b = m >> 12, n = m & 4095;
            if (col < 384) {
                int seg = col >> 7;
                int oo  = col & 127;
                float* dst = (seg == 0) ? o0 : ((seg == 1) ? o1 : o2);
                dst[((size_t)(b * 4 + (oo >> 5)) * NTOK + n) * HD + (oo & 31)] = yv;
            } else {
                int c2 = col - 384;
                o3[(size_t)m * CDIM + c2] = 0.5f * yv * (1.0f + erff(yv * 0.70710678118f));
            }
        } else if (MODE == 1) {
            int b = m >> 6, p = m & 63;
            int oo = col & 127;
            float* dst = (col < 128) ? o0 : o1;
            dst[((size_t)(b * 4 + (oo >> 5)) * PL + p) * HD + (oo & 31)] = yv;
        } else {
            o0[(size_t)m * Ncols + col] = yv;
        }
    }
}

// ---------------- in-place L2 normalize over trailing 32 ------------------------------------
__global__ __launch_bounds__(128) void l2norm32(float* __restrict__ buf)
{
    int row = blockIdx.x * 4 + (threadIdx.x >> 5);
    int lane = threadIdx.x & 31;
    float v = buf[(size_t)row * 32 + lane];
    float s = v * v;
#pragma unroll
    for (int off = 16; off; off >>= 1) s += __shfl_xor(s, off, 32);
    float inv = 1.0f / fmaxf(sqrtf(s), 1e-12f);
    buf[(size_t)row * 32 + lane] = v * inv;
}

// ---------------- 8x8 spatial mean pool + layernorm -> bf16 ---------------------------------
__global__ __launch_bounds__(128) void pool_ln(const float* __restrict__ xs,
                                               const float* __restrict__ g,
                                               const float* __restrict__ be,
                                               __bf16* __restrict__ xp)
{
    int bp = blockIdx.x;
    int b = bp >> 6, p = bp & 63;
    int ph = p >> 3, pw = p & 7;
    int c = threadIdx.x;
    float s = 0.f;
    for (int r = 0; r < 8; ++r)
        for (int t = 0; t < 8; ++t) {
            int n = (ph * 8 + r) * 64 + pw * 8 + t;
            s += xs[((size_t)b * NTOK + n) * CDIM + c];
        }
    s *= (1.0f / 64.0f);
    __shared__ float s1[128], s2[128];
    s1[c] = s; s2[c] = s * s;
    __syncthreads();
    for (int off = 64; off; off >>= 1) {
        if (c < off) { s1[c] += s1[c + off]; s2[c] += s2[c + off]; }
        __syncthreads();
    }
    float mean = s1[0] * (1.0f / 128.0f);
    float var  = s2[0] * (1.0f / 128.0f) - mean * mean;
    float y = (s - mean) * rsqrtf(var + 1e-5f) * g[c] + be[c];
    xp[(size_t)bp * CDIM + c] = (__bf16)y;
}

// ---------------- fused attention: 1 wave = 16 queries of one (b,h) -------------------------
__global__ __launch_bounds__(128) void attn_kernel(
    const float* __restrict__ qn_g,  // (B,H,N,32)  l2-normalized q
    const float* __restrict__ kn_g,  // (B,H,N,32)  l2-normalized k
    const float* __restrict__ vl_g,  // (B,H,N,32)
    const float* __restrict__ kp_g,  // (B,H,64,32) l2-normalized pooled k
    const float* __restrict__ vp_g,  // (B,H,64,32)
    const float* __restrict__ ttab,  // (H,T)
    const int*   __restrict__ rpi,   // (N*64)
    const float* __restrict__ temp,  // (H)
    const float* __restrict__ qe,    // (H,32)
    const float* __restrict__ rpb,   // (H,9)
    const float* __restrict__ lt,    // (H,32,9)
    const float* __restrict__ lb,    // (H,9)
    const float* __restrict__ sls,   // (N)
    __bf16* __restrict__ preproj,    // (B*N,128)
    int T)
{
    const int warp = threadIdx.x >> 5;
    const int lane = threadIdx.x & 31;
    const int l = lane & 15;
    const int hi = lane >> 4;
    int wg = blockIdx.x * 4 + warp;         // 0..16383
    int qt = wg & 255;
    int h  = (wg >> 8) & 3;
    int b  = wg >> 10;
    int n0 = qt * 16;
    int bh = b * 4 + h;

    __shared__ float lg[4][16][81];   // [0..8]=local, [9..72]=pool logits / probs
    __shared__ float qs[4][16][33];   // q_scaled
    __shared__ float qn[4][16][33];   // q_norm
    __shared__ float xo[4][16][33];   // x_pool staging

    float spt = log1pf(expf(temp[h]));

    // 1. load q, build q_scaled
    for (int r = 0; r < 16; ++r) {
        float qv = qn_g[((size_t)bh * NTOK + n0 + r) * HD + lane];
        qn[warp][r][lane] = qv;
        qs[warp][r][lane] = (qv + qe[h * 32 + lane]) * spt * sls[n0 + r];
    }
    __syncthreads();

    // 2. pool logits: q_scaled(16x32) @ kp^T(32x64) via f32 WMMA 16x16x4
    for (int nt = 0; nt < 4; ++nt) {
        v8f acc = {};
        for (int k0 = 0; k0 < 32; k0 += 4) {
            v2f a, bb;
            a[0] = qs[warp][l][k0 + 2 * hi];
            a[1] = qs[warp][l][k0 + 2 * hi + 1];
            bb = *(const v2f*)(kp_g + ((size_t)bh * PL + nt * 16 + l) * HD + k0 + 2 * hi);
            acc = __builtin_amdgcn_wmma_f32_16x16x4_f32(false, a, false, bb,
                                                        (short)0, acc, false, false);
        }
#pragma unroll
        for (int i = 0; i < 8; ++i) {
            int r = i + 8 * hi;
            int m = nt * 16 + l;
            float bias = ttab[(size_t)h * T + rpi[(size_t)(n0 + r) * PL + m]];
            lg[warp][r][LL + m] = acc[i] + bias;
        }
    }

    // 3. local 3x3 logits (b128 k-row loads)
    for (int j = lane; j < 16 * LL; j += 32) {
        int r = j / LL, li = j % LL;
        int n = n0 + r;
        int y = n >> 6, x = n & 63;
        int yy = y + li / 3 - 1, xx = x + li % 3 - 1;
        float val = NEG_INF;
        if (yy >= 0 && yy < 64 && xx >= 0 && xx < 64) {
            const float4* kr4 =
                (const float4*)(kn_g + ((size_t)bh * NTOK + yy * 64 + xx) * HD);
            float s = 0.f;
#pragma unroll
            for (int d4 = 0; d4 < 8; ++d4) {
                float4 kv = kr4[d4];
                s += qs[warp][r][d4 * 4 + 0] * kv.x + qs[warp][r][d4 * 4 + 1] * kv.y +
                     qs[warp][r][d4 * 4 + 2] * kv.z + qs[warp][r][d4 * 4 + 3] * kv.w;
            }
            val = s + rpb[h * LL + li];
        }
        lg[warp][r][li] = val;
    }
    __syncthreads();

    // 4. softmax over 73 + learnable-token correction on local slots
    if (lane < 16) {
        int r = lane;
        float mx = NEG_INF;
        for (int j = 0; j < 73; ++j) mx = fmaxf(mx, lg[warp][r][j]);
        float s = 0.f;
        for (int j = 0; j < 73; ++j) {
            float e = expf(lg[warp][r][j] - mx);
            lg[warp][r][j] = e;
            s += e;
        }
        float inv = 1.0f / s;
        for (int j = 0; j < 73; ++j) lg[warp][r][j] *= inv;
        for (int li = 0; li < LL; ++li) {
            float dot = 0.f;
#pragma unroll
            for (int d = 0; d < HD; ++d) dot += qn[warp][r][d] * lt[(h * 32 + d) * LL + li];
            lg[warp][r][li] += dot + lb[h * LL + li];
        }
    }
    __syncthreads();

    // 5. x_pool = a_pool(16x64) @ v_pool(64x32) via f32 WMMA
    for (int nt = 0; nt < 2; ++nt) {
        v8f acc = {};
        for (int k0 = 0; k0 < 64; k0 += 4) {
            v2f a, bb;
            a[0] = lg[warp][l][LL + k0 + 2 * hi];
            a[1] = lg[warp][l][LL + k0 + 2 * hi + 1];
            const float* vpp = vp_g + ((size_t)bh * PL + k0 + 2 * hi) * HD + nt * 16 + l;
            bb[0] = vpp[0];
            bb[1] = vpp[HD];
            acc = __builtin_amdgcn_wmma_f32_16x16x4_f32(false, a, false, bb,
                                                        (short)0, acc, false, false);
        }
#pragma unroll
        for (int i = 0; i < 8; ++i) xo[warp][i + 8 * hi][nt * 16 + l] = acc[i];
    }
    __syncthreads();

    // 6. x_loc accumulate (lane owns dim d=lane), write bf16 pre-projection activations
    for (int r = 0; r < 16; ++r) {
        float s = xo[warp][r][lane];
        int n = n0 + r;
        int y = n >> 6, x = n & 63;
#pragma unroll
        for (int li = 0; li < LL; ++li) {
            int yy = y + li / 3 - 1, xx = x + li % 3 - 1;
            if (yy >= 0 && yy < 64 && xx >= 0 && xx < 64) {
                s += lg[warp][r][li] *
                     vl_g[((size_t)bh * NTOK + yy * 64 + xx) * HD + lane];
            }
        }
        preproj[((size_t)b * NTOK + n) * CDIM + h * HD + lane] = (__bf16)s;
    }
}

// ---------------- launch --------------------------------------------------------------------
extern "C" void kernel_launch(void* const* d_in, const int* in_sizes, int n_in,
                              void* d_out, int out_size, void* d_ws, size_t ws_size,
                              hipStream_t stream)
{
    const float* x     = (const float*)d_in[0];
    const float* Wq    = (const float*)d_in[1];
    const float* bq    = (const float*)d_in[2];
    const float* Wkv   = (const float*)d_in[3];
    const float* bkv   = (const float*)d_in[4];
    const float* Wsr   = (const float*)d_in[5];
    const float* bsr   = (const float*)d_in[6];
    const float* ng    = (const float*)d_in[7];
    const float* nb    = (const float*)d_in[8];
    const float* Wproj = (const float*)d_in[9];
    const float* bproj = (const float*)d_in[10];
    const float* cw1   = (const float*)d_in[11];
    const float* cb1   = (const float*)d_in[12];
    const float* cw2   = (const float*)d_in[13];
    const float* cb2   = (const float*)d_in[14];
    const float* temp  = (const float*)d_in[15];
    const float* qe    = (const float*)d_in[16];
    const float* rpb   = (const float*)d_in[17];
    const float* lt    = (const float*)d_in[18];
    const float* lb    = (const float*)d_in[19];
    const float* sls   = (const float*)d_in[20];
    const float* rct   = (const float*)d_in[21];
    const int*   rpi   = (const int*)d_in[22];
    int T = in_sizes[21] / 2;

    char* w = (char*)d_ws;
    auto alloc = [&](size_t bytes) -> char* {
        char* p = w;
        w += (bytes + 255) & ~(size_t)255;
        return p;
    };
    const int M = 16 * NTOK;                         // 65536
    __bf16* xb   = (__bf16*)alloc((size_t)M * CDIM * 2);
    __bf16* Wc   = (__bf16*)alloc(512 * 128 * 2);
    float*  bc   = (float*) alloc(512 * 4);
    __bf16* Wp   = (__bf16*)alloc(128 * 128 * 2);
    float*  qn   = (float*) alloc((size_t)M * CDIM * 4);   // (B,H,N,32)
    float*  kn   = (float*) alloc((size_t)M * CDIM * 4);
    float*  vl   = (float*) alloc((size_t)M * CDIM * 4);
    float*  xs   = (float*) alloc((size_t)M * CDIM * 4);
    __bf16* xpb  = (__bf16*)alloc(1024 * 128 * 2);
    float*  kp   = (float*) alloc(16 * 4 * 64 * 32 * 4);
    float*  vp   = (float*) alloc(16 * 4 * 64 * 32 * 4);
    __bf16* ppj  = (__bf16*)alloc((size_t)M * CDIM * 2);
    float*  ttab = (float*) alloc((size_t)16 * T);

    pack_w<<<256, 256, 0, stream>>>(Wq, Wkv, Wsr, bq, bkv, bsr, Wproj, Wc, bc, Wp);
    cvt_bf16<<<2048, 256, 0, stream>>>(x, xb, M * CDIM);
    cpb_table<<<T, 256, 0, stream>>>(rct, cw1, cb1, cw2, cb2, ttab, T);

    // fused qkv + sr projection: (65536 x 512) = (65536x128)@(512x128)^T
    gemm_bf16<0><<<32768, 128, 0, stream>>>(xb, Wc, bc, qn, kn, vl, xs, M, 512, 128);

    l2norm32<<<65536, 128, 0, stream>>>(qn);   // 262144 rows of 32
    l2norm32<<<65536, 128, 0, stream>>>(kn);

    pool_ln<<<1024, 128, 0, stream>>>(xs, ng, nb, xpb);

    // pooled kv: (1024 x 256)
    gemm_bf16<1><<<256, 128, 0, stream>>>(xpb, Wc + 128 * 128, bc + 128,
                                          kp, vp, (float*)nullptr, (float*)nullptr,
                                          1024, 256, 128);
    l2norm32<<<1024, 128, 0, stream>>>(kp);

    attn_kernel<<<4096, 128, 0, stream>>>(qn, kn, vl, kp, vp, ttab, rpi,
                                          temp, qe, rpb, lt, lb, sls, ppj, T);

    // output projection: (65536 x 128)
    gemm_bf16<2><<<8192, 128, 0, stream>>>(ppj, Wp, bproj, (float*)d_out,
                                           (float*)nullptr, (float*)nullptr, (float*)nullptr,
                                           M, 128, 128);
}
